// Attention_14817637171478
// MI455X (gfx1250) — matrix operand order repaired and checked by reading, not compiled, not run on hardware
//
#include <hip/hip_runtime.h>

#define D_MODEL 1024
#define HEADS 16
#define HEAD_DIM 64
#define BATCH 4
#define SEQ 2048
#define MROWS (BATCH * SEQ)   // 8192

// LDS B-tile row stride for the 64x64 staged tile: 72 bf16 = 144 B.
#define BSTRIDE2 72
// Fused-attention LDS probs row stride: 2048 + 4 pad floats.
// 16B aligned rows (8208 B); start-bank = 4*row mod 64 -> conflict-free b128.
#define PROW 2052

typedef __attribute__((ext_vector_type(16))) __bf16 v16bf;
typedef __attribute__((ext_vector_type(8)))  __bf16 v8bf;
typedef __attribute__((ext_vector_type(8)))  float  v8f;
typedef __attribute__((ext_vector_type(4)))  float  v4f;
typedef __attribute__((ext_vector_type(4)))  int    v4i;

typedef __attribute__((address_space(1))) v4i g_v4i;   // global
typedef __attribute__((address_space(3))) v4i l_v4i;   // LDS

#if defined(__gfx1250__) && __has_builtin(__builtin_amdgcn_global_load_async_to_lds_b128)
#define USE_ASYNC_LDS 1
#else
#define USE_ASYNC_LDS 0
#endif

__device__ __forceinline__ void async_copy16(const __bf16* src, __bf16* dst) {
#if USE_ASYNC_LDS
  __builtin_amdgcn_global_load_async_to_lds_b128(
      (g_v4i*)(__bf16*)src, (l_v4i*)dst, 0, 0);   // global_load_async_to_lds_b128
#else
  *(v8bf*)dst = *(const v8bf*)src;
#endif
}

__device__ __forceinline__ void async_stage_fence() {
#if USE_ASYNC_LDS
  asm volatile("s_wait_asynccnt 0x0" ::: "memory");
#endif
}

// ---------------------------------------------------------------------------
// WMMA tile loaders (CDNA5 wave32 layouts, CDNA5 ISA §7.12.2)
// ---------------------------------------------------------------------------
__device__ __forceinline__ v16bf load_tileA_bf16(const __bf16* __restrict__ A, int ldk) {
  int lane = threadIdx.x & 31;
  int row  = lane & 15;
  int koff = (lane >> 4) * 8;
  const __bf16* p0 = A + (size_t)row * ldk + koff;
  v8bf lo = *(const v8bf*)(p0);
  v8bf hi = *(const v8bf*)(p0 + 16);
  v16bf r;
#pragma unroll
  for (int i = 0; i < 8; ++i) { r[i] = lo[i]; r[i + 8] = hi[i]; }
  return r;
}

// A tile from f32 memory (LDS probs or global), converted to bf16 on load
__device__ __forceinline__ v16bf load_tileA_f32(const float* A, int lda) {
  int lane = threadIdx.x & 31;
  int row  = lane & 15;
  int koff = (lane >> 4) * 8;
  const float* p0 = A + (size_t)row * lda + koff;
  v8f lo = *(const v8f*)(p0);
  v8f hi = *(const v8f*)(p0 + 16);
  v16bf r;
#pragma unroll
  for (int i = 0; i < 8; ++i) { r[i] = (__bf16)lo[i]; r[i + 8] = (__bf16)hi[i]; }
  return r;
}

// B tile (32x16) from row-major [N,K] memory: columns are contiguous rows.
__device__ __forceinline__ v16bf load_tileB_global(const __bf16* __restrict__ BT, int ldk) {
  int lane = threadIdx.x & 31;
  const __bf16* p = BT + (size_t)(lane & 15) * ldk + (lane >> 4) * 8;
  v8bf lo = *(const v8bf*)(p);
  v8bf hi = *(const v8bf*)(p + 16);
  v16bf r;
#pragma unroll
  for (int i = 0; i < 8; ++i) { r[i] = lo[i]; r[i + 8] = hi[i]; }
  return r;
}

struct ATiles { v16bf a[2]; };

__device__ __forceinline__ ATiles loadA2_bf16(const __bf16* __restrict__ A, int ldk) {
  ATiles at;
  at.a[0] = load_tileA_bf16(A, ldk);
  at.a[1] = load_tileA_bf16(A + 32, ldk);
  return at;
}

// Cooperative stage of a 64(col) x 64(K) bf16 B tile into LDS (async).
__device__ __forceinline__ void stage_B2(const __bf16* __restrict__ BT, int ldk,
                                         __bf16* buf) {
  int t = threadIdx.x;
  int c = t >> 2;
  int j = (t & 3) * 8;
  const __bf16* src = BT + (size_t)c * ldk + j;
  __bf16* dst = buf + c * BSTRIDE2 + j;
  async_copy16(src, dst);
  async_copy16(src + 32, dst + 32);
}

__device__ __forceinline__ v16bf lds_tileB2(const __bf16* __restrict__ buf,
                                            int col16, int kk) {
  int lane = threadIdx.x & 31;
  int col  = col16 + (lane & 15);
  int koff = kk + (lane >> 4) * 8;
  const __bf16* p = buf + col * BSTRIDE2 + koff;
  v8bf lo = *(const v8bf*)(p);
  v8bf hi = *(const v8bf*)(p + 16);
  v16bf r;
#pragma unroll
  for (int i = 0; i < 8; ++i) { r[i] = lo[i]; r[i + 8] = hi[i]; }
  return r;
}

__device__ __forceinline__ void gemm_phase(const __bf16* __restrict__ buf,
                                           const ATiles& at, v8f (&acc)[4]) {
#pragma unroll
  for (int kk = 0; kk < 2; ++kk) {
    v16bf b0 = lds_tileB2(buf, 0,  kk * 32);
    v16bf b1 = lds_tileB2(buf, 16, kk * 32);
    v16bf b2 = lds_tileB2(buf, 32, kk * 32);
    v16bf b3 = lds_tileB2(buf, 48, kk * 32);
    acc[0] = __builtin_amdgcn_wmma_f32_16x16x32_bf16(false, at.a[kk], false, b0,
                                                     (short)0, acc[0], false, false);
    acc[1] = __builtin_amdgcn_wmma_f32_16x16x32_bf16(false, at.a[kk], false, b1,
                                                     (short)0, acc[1], false, false);
    acc[2] = __builtin_amdgcn_wmma_f32_16x16x32_bf16(false, at.a[kk], false, b2,
                                                     (short)0, acc[2], false, false);
    acc[3] = __builtin_amdgcn_wmma_f32_16x16x32_bf16(false, at.a[kk], false, b3,
                                                     (short)0, acc[3], false, false);
  }
}

// ---------------------------------------------------------------------------
// Elementwise f32 -> bf16 conversion (vectorized x4)
// ---------------------------------------------------------------------------
__global__ __launch_bounds__(256) void cvt_f32_bf16_kernel(const float* __restrict__ in,
                                                           __bf16* __restrict__ out,
                                                           size_t n4) {
  size_t i = ((size_t)blockIdx.x * blockDim.x + threadIdx.x);
  if (i >= n4) return;
  size_t idx = i * 4;
  v4f v = *(const v4f*)(in + idx);
#pragma unroll
  for (int j = 0; j < 4; ++j) out[idx + j] = (__bf16)v[j];
}

// W [K, N] f32 row-major  ->  WT [N, K] bf16 row-major (LDS tiled transpose)
__global__ __launch_bounds__(256) void transpose_cvt_kernel(const float* __restrict__ W,
                                                            __bf16* __restrict__ WT) {
  __shared__ float tile[32][33];
  int n0 = blockIdx.x * 32;
  int k0 = blockIdx.y * 32;
  int tx = threadIdx.x & 31;
  int ty = threadIdx.x >> 5;
#pragma unroll
  for (int i = 0; i < 32; i += 8)
    tile[ty + i][tx] = W[(size_t)(k0 + ty + i) * D_MODEL + (n0 + tx)];
  __syncthreads();
#pragma unroll
  for (int i = 0; i < 32; i += 8)
    WT[(size_t)(n0 + ty + i) * D_MODEL + (k0 + tx)] = (__bf16)tile[tx][ty + i];
}

// ---------------------------------------------------------------------------
// Kernel 1: QKV projection (unchanged from round 5; async-LDS double buffer).
// ---------------------------------------------------------------------------
__global__ __launch_bounds__(256) void qkv_proj_kernel(
    const __bf16* __restrict__ qbf, const __bf16* __restrict__ kbf,
    const __bf16* __restrict__ vbf, const __bf16* __restrict__ WqT,
    const __bf16* __restrict__ WkT, const __bf16* __restrict__ WvT,
    __bf16* __restrict__ Qh, __bf16* __restrict__ Kh, __bf16* __restrict__ VhT) {
  __shared__ __bf16 bufB[2][64 * BSTRIDE2];

  const int mode = blockIdx.z;
  const __bf16* X  = (mode == 0) ? qbf : (mode == 1) ? kbf : vbf;
  const __bf16* WT = (mode == 0) ? WqT : (mode == 1) ? WkT : WvT;

  const int wave = threadIdx.x >> 5;
  const int m0 = blockIdx.x * 128 + wave * 16;
  const int n0 = blockIdx.y * 64;
  const __bf16* Abase = X + (size_t)m0 * D_MODEL;
  const __bf16* Bbase = WT + (size_t)n0 * D_MODEL;

  v8f acc[4] = {v8f{}, v8f{}, v8f{}, v8f{}};

  stage_B2(Bbase, D_MODEL, bufB[0]);
  ATiles a0 = loadA2_bf16(Abase, D_MODEL);
  ATiles a1;
  async_stage_fence();
  __syncthreads();

  const int KT = D_MODEL / 64;  // 16
  for (int kt = 0; kt < KT; kt += 2) {
    stage_B2(Bbase + (kt + 1) * 64, D_MODEL, bufB[1]);
    a1 = loadA2_bf16(Abase + (kt + 1) * 64, D_MODEL);
    gemm_phase(bufB[0], a0, acc);
    async_stage_fence();
    __syncthreads();
    if (kt + 2 < KT) {
      stage_B2(Bbase + (kt + 2) * 64, D_MODEL, bufB[0]);
      a0 = loadA2_bf16(Abase + (kt + 2) * 64, D_MODEL);
    }
    gemm_phase(bufB[1], a1, acc);
    async_stage_fence();
    __syncthreads();
  }

  const int lane  = threadIdx.x & 31;
  const int nloc  = lane & 15;
  const int mbase = (lane >> 4) * 8;
#pragma unroll
  for (int t = 0; t < 4; ++t) {
#pragma unroll
    for (int r = 0; r < 8; ++r) {
      int gm = m0 + mbase + r;
      int gn = n0 + t * 16 + nloc;
      int b = gm >> 11, s = gm & (SEQ - 1);
      int h = gn >> 6,  d = gn & (HEAD_DIM - 1);
      __bf16 val = (__bf16)acc[t][r];
      if (mode == 0)
        Qh[((size_t)(b * HEADS + h) * SEQ + s) * HEAD_DIM + d] = val;
      else if (mode == 1)
        Kh[((size_t)(b * HEADS + h) * SEQ + s) * HEAD_DIM + d] = val;
      else
        VhT[((size_t)(b * HEADS + h) * HEAD_DIM + d) * SEQ + s] = val;
    }
  }
}

// ---------------------------------------------------------------------------
// Kernel 2 (FUSED): scores -> alibi+mask -> softmax -> a-store -> A@V.
// Block: 32 sq rows x full 2048 sk, probs resident in LDS (256.5 KB of the
// WGP's 320 KB). `a` is written to HBM exactly once and never re-read.
// 8 waves: wave w -> m-tile (w>>2)*16, n/d-tile (w&3)*16.
// ---------------------------------------------------------------------------
__global__ __launch_bounds__(256) void attn_fused_kernel(
    const __bf16* __restrict__ Qh, const __bf16* __restrict__ Kh,
    const __bf16* __restrict__ VhT, const float* __restrict__ alibi,
    const unsigned char* __restrict__ mask,
    float* __restrict__ a_out, float* __restrict__ out) {
  __shared__ float probs[32 * PROW];   // 262,656 B
  __shared__ float red[256];
  __shared__ float rowMax[32];
  __shared__ float rowInv[32];

  const int bh = blockIdx.z;
  const int b = bh >> 4, h = bh & (HEADS - 1);
  const int sq0 = blockIdx.x * 32;
  const int wave = threadIdx.x >> 5;
  const int lane = threadIdx.x & 31;
  const int mt = (wave >> 2) * 16;   // 0 / 16
  const int nt = (wave & 3) * 16;    // 0/16/32/48

  const __bf16* Qb = Qh  + (size_t)bh * SEQ * HEAD_DIM;
  const __bf16* Kb = Kh  + (size_t)bh * SEQ * HEAD_DIM;
  const __bf16* Vb = VhT + (size_t)bh * HEAD_DIM * SEQ;

  const int nloc  = lane & 15;
  const int mbase = (lane >> 4) * 8;
  const float NEG_INF = -__builtin_inff();

  // Q tiles for this wave's 16 rows (K=64 -> two 16x32 tiles), loaded once.
  v16bf aq0 = load_tileA_bf16(Qb + (size_t)(sq0 + mt) * HEAD_DIM, HEAD_DIM);
  v16bf aq1 = load_tileA_bf16(Qb + (size_t)(sq0 + mt) * HEAD_DIM + 32, HEAD_DIM);

  // ---- Phase A: raw scores / 8 -> LDS (each wave: one 16x16 tile / 64 cols)
  for (int s64 = 0; s64 < SEQ; s64 += 64) {
    const int sk0 = s64 + nt;
    v8f acc = v8f{};
#pragma unroll
    for (int k = 0; k < HEAD_DIM; k += 32) {
      v16bf bm = load_tileB_global(Kb + (size_t)sk0 * HEAD_DIM + k, HEAD_DIM);
      acc = __builtin_amdgcn_wmma_f32_16x16x32_bf16(false, (k ? aq1 : aq0), false, bm,
                                                    (short)0, acc, false, false);
    }
    float* prow = probs + (size_t)(mt + mbase) * PROW + sk0 + nloc;
#pragma unroll
    for (int r = 0; r < 8; ++r) prow[(size_t)r * PROW] = acc[r] * 0.125f;
  }
  __syncthreads();

  // ---- Phase B: alibi + mask + softmax in LDS; coalesced store of `a`
  {
    const float* arow = alibi + (size_t)h * SEQ * SEQ + (size_t)sq0 * SEQ;
    const int r = threadIdx.x >> 3;   // 0..31 (row)
    const int p = threadIdx.x & 7;    // 0..7  (column residue)

    float mx = NEG_INF;
    for (int j = 0; j < 256; ++j) {
      int c = p + 8 * j;
      float x = probs[r * PROW + c] + arow[(size_t)r * SEQ + c];
      if (mask[b * SEQ + c]) x = NEG_INF;
      probs[r * PROW + c] = x;
      mx = fmaxf(mx, x);
    }
    red[threadIdx.x] = mx;
    __syncthreads();
    if (threadIdx.x < 32) {
      float m = red[threadIdx.x * 8];
#pragma unroll
      for (int i = 1; i < 8; ++i) m = fmaxf(m, red[threadIdx.x * 8 + i]);
      if (m == NEG_INF) m = 0.0f;   // fully masked row guard
      rowMax[threadIdx.x] = m;
    }
    __syncthreads();

    float m = rowMax[r];
    float s = 0.0f;
    for (int j = 0; j < 256; ++j) {
      int c = p + 8 * j;
      float e = __expf(probs[r * PROW + c] - m);
      probs[r * PROW + c] = e;
      s += e;
    }
    red[threadIdx.x] = s;
    __syncthreads();
    if (threadIdx.x < 32) {
      float t = 0.0f;
#pragma unroll
      for (int i = 0; i < 8; ++i) t += red[threadIdx.x * 8 + i];
      rowInv[threadIdx.x] = (t > 0.0f) ? 1.0f / t : 0.0f;
    }
    __syncthreads();

    // normalize in LDS + single coalesced write of `a` to HBM
    float* grow = a_out + (size_t)bh * SEQ * SEQ + (size_t)sq0 * SEQ;
    for (int r2 = 0; r2 < 32; ++r2) {
      float inv = rowInv[r2];
#pragma unroll
      for (int j = 0; j < 8; ++j) {
        int c = threadIdx.x + j * 256;
        float val = probs[r2 * PROW + c] * inv;
        probs[r2 * PROW + c] = val;
        grow[(size_t)r2 * SEQ + c] = val;
      }
    }
  }
  __syncthreads();

  // ---- Phase C: out = probs @ V  (A from LDS f32->bf16, B from L2 global)
  v8f acc = v8f{};
  for (int k = 0; k < SEQ; k += 32) {
    v16bf am = load_tileA_f32(probs + (size_t)mt * PROW + k, PROW);  // ds_load
    v16bf bm = load_tileB_global(Vb + (size_t)nt * SEQ + k, SEQ);
    acc = __builtin_amdgcn_wmma_f32_16x16x32_bf16(false, am, false, bm,
                                                  (short)0, acc, false, false);
  }
  const int d = nt + nloc;
#pragma unroll
  for (int r = 0; r < 8; ++r) {
    int sq = sq0 + mt + mbase + r;
    out[((size_t)b * SEQ + sq) * D_MODEL + h * HEAD_DIM + d] = acc[r];
  }
}

// ---------------------------------------------------------------------------
// Host launcher
// ---------------------------------------------------------------------------
extern "C" void kernel_launch(void* const* d_in, const int* in_sizes, int n_in,
                              void* d_out, int out_size, void* d_ws, size_t ws_size,
                              hipStream_t stream) {
  const float* q = (const float*)d_in[0];
  const float* k = (const float*)d_in[1];
  const float* v = (const float*)d_in[2];
  const unsigned char* mask = (const unsigned char*)d_in[3];  // jnp.bool_ (1 byte)
  const float* alibi = (const float*)d_in[4];
  const float* Wq = (const float*)d_in[5];
  const float* Wk = (const float*)d_in[6];
  const float* Wv = (const float*)d_in[7];

  float* out   = (float*)d_out;
  float* a_out = out + (size_t)BATCH * SEQ * D_MODEL;  // `a` region

  char* ws = (char*)d_ws;
  size_t off = 0;
  auto carve = [&](size_t elems) {
    void* p = ws + off;
    off += (elems * sizeof(__bf16) + 255) & ~(size_t)255;
    return (__bf16*)p;
  };
  const size_t NX = (size_t)MROWS * D_MODEL;
  const size_t NW = (size_t)D_MODEL * D_MODEL;
  __bf16* qbf = carve(NX);
  __bf16* kbf = carve(NX);
  __bf16* vbf = carve(NX);
  __bf16* WqT = carve(NW);
  __bf16* WkT = carve(NW);
  __bf16* WvT = carve(NW);
  __bf16* Qh  = carve(NX);   // [B,H,S,64]
  __bf16* Kh  = carve(NX);   // [B,H,S,64]
  __bf16* VhT = carve(NX);   // [B,H,64,S]
  (void)ws_size; (void)in_sizes; (void)n_in; (void)out_size;

  const dim3 blk(256);

  {
    size_t n4 = NX / 4;
    dim3 grid((unsigned)((n4 + 255) / 256));
    cvt_f32_bf16_kernel<<<grid, blk, 0, stream>>>(q, qbf, n4);
    cvt_f32_bf16_kernel<<<grid, blk, 0, stream>>>(k, kbf, n4);
    cvt_f32_bf16_kernel<<<grid, blk, 0, stream>>>(v, vbf, n4);
  }
  {
    dim3 grid(D_MODEL / 32, D_MODEL / 32);
    transpose_cvt_kernel<<<grid, blk, 0, stream>>>(Wq, WqT);
    transpose_cvt_kernel<<<grid, blk, 0, stream>>>(Wk, WkT);
    transpose_cvt_kernel<<<grid, blk, 0, stream>>>(Wv, WvT);
  }

  // 1) QKV projections
  qkv_proj_kernel<<<dim3(MROWS / 128, D_MODEL / 64, 3), blk, 0, stream>>>(
      qbf, kbf, vbf, WqT, WkT, WvT, Qh, Kh, VhT);

  // 2) fused scores+softmax+AV: writes `a` once, out once
  attn_fused_kernel<<<dim3(SEQ / 32, 1, BATCH * HEADS), blk, 0, stream>>>(
      Qh, Kh, VhT, alibi, mask, a_out, out);
}
